// GAT_15865609192051
// MI455X (gfx1250) — compile-verified, matrix operand order checked by reading, not run on hardware
//
#include <hip/hip_runtime.h>
#include <hip/hip_bf16.h>

#define Nn 4096
#define Ff 512
#define Hh 256
#define Kk 8
#define KH 2048   // Kk*Hh
#define NEGBIG -1e9f
#define LDSROW 40          // padded row stride (halves) for 256x32 bf16 slab

typedef __attribute__((ext_vector_type(16))) __bf16 v16bf;
typedef __attribute__((ext_vector_type(8)))  __bf16 v8bf;
typedef __attribute__((ext_vector_type(8)))  float  v8f;
typedef __attribute__((ext_vector_type(4)))  float  v4f;

// ---------------------------------------------------------------------------
// Fragment loader: 16-bit 16x32 A/B layout (ISA 7.12.2).
// Element e of lane L holds (row r0 + L%16, col c0 + (e%8) + 8*(L/16) + 16*(e/8)).
// Works for A (row-major [M x Kd]) and for B supplied TRANSPOSED ([Nd x Kd]).
// ---------------------------------------------------------------------------
__device__ __forceinline__ v16bf load_frag(const __bf16* base, int ld, int r0, int c0, int lane) {
    int m = r0 + (lane & 15);
    int b = lane >> 4;
    const __bf16* p = base + (size_t)m * ld + c0 + 8 * b;
    v8bf lo = *(const v8bf*)p;
    v8bf hi = *(const v8bf*)(p + 16);
    v16bf r;
#pragma unroll
    for (int e = 0; e < 8; ++e) { r[e] = lo[e]; r[e + 8] = hi[e]; }
    return r;
}

__device__ __forceinline__ v8f wmma_bf16(v16bf a, v16bf b, v8f c) {
    return __builtin_amdgcn_wmma_f32_16x16x32_bf16(false, a, false, b, (short)0, c, false, false);
}

// Async stage: thread `tid` copies row `tid` (32 halves) of gsrc[.., j0..j0+31]
// into LDS slab row tid (padded stride LDSROW), as 4x b128 async transfers.
#define ASYNC_STAGE(gsrc, ldg, j0, lbuf_elem_ptr)                                   \
    do {                                                                            \
        const __bf16* _g = (gsrc) + (size_t)threadIdx.x * (ldg) + (j0);             \
        unsigned _l = (unsigned)(uintptr_t)(lbuf_elem_ptr);                         \
        _Pragma("unroll")                                                           \
        for (int _c = 0; _c < 4; ++_c) {                                            \
            asm volatile("global_load_async_to_lds_b128 %0, %1, off"                \
                         :: "v"(_l + _c * 16), "v"(_g + _c * 8) : "memory");        \
        }                                                                           \
    } while (0)

// LDS fragment read (same element mapping as load_frag, row stride LDSROW).
__device__ __forceinline__ v16bf lds_frag(const __bf16* lbuf, int t, int lane) {
    int n = 16 * t + (lane & 15);
    int b = lane >> 4;
    const __bf16* p = lbuf + n * LDSROW + 8 * b;
    v8bf lo = *(const v8bf*)p;
    v8bf hi = *(const v8bf*)(p + 16);
    v16bf r;
#pragma unroll
    for (int e = 0; e < 8; ++e) { r[e] = lo[e]; r[e + 8] = hi[e]; }
    return r;
}

// ---------------------------------------------------------------------------
// Elementwise fp32 -> bf16
// ---------------------------------------------------------------------------
__global__ void k_cvt(const float* __restrict__ in, __bf16* __restrict__ out, size_t n) {
    size_t i = (size_t)blockIdx.x * blockDim.x + threadIdx.x;
    if (i < n) out[i] = (__bf16)in[i];
}

// Transpose + convert: out[c*R + r] = in[r*C + c]   (in is [R x C])
__global__ void k_cvtT(const float* __restrict__ in, __bf16* __restrict__ out, int R, int C) {
    size_t i = (size_t)blockIdx.x * blockDim.x + threadIdx.x;
    size_t tot = (size_t)R * C;
    if (i >= tot) return;
    int r = (int)(i / C), c = (int)(i % C);
    out[(size_t)c * R + r] = (__bf16)in[i];
}

// ---------------------------------------------------------------------------
// GEMM: C[M x Nd] = A[M x Kd](bf16) * BT[Nd x Kd]^T (bf16) + bias
// one wave per 16x64 output block; optional fp32 and/or bf16 outputs.
// ---------------------------------------------------------------------------
__global__ void k_gemm(const __bf16* __restrict__ A, const __bf16* __restrict__ BT,
                       const float* __restrict__ bias,
                       float* __restrict__ Cf, __bf16* __restrict__ Cb,
                       int M, int Nd, int Kd) {
    int wave = (int)((blockIdx.x * blockDim.x + threadIdx.x) >> 5);
    int lane = threadIdx.x & 31;
    int cgrp = Nd / 64;
    int rt = wave / cgrp, cg = wave % cgrp;
    if (rt * 16 >= M) return;
    int m0 = rt * 16, n0 = cg * 64;

    v8f acc[4] = {};
    for (int k0 = 0; k0 < Kd; k0 += 32) {
        v16bf af = load_frag(A, Kd, m0, k0, lane);
#pragma unroll
        for (int t = 0; t < 4; ++t) {
            v16bf bfr = load_frag(BT, Kd, n0 + 16 * t, k0, lane);
            acc[t] = wmma_bf16(af, bfr, acc[t]);
        }
    }
    int mbase = m0 + 8 * (lane >> 4);
    int ncol  = lane & 15;
#pragma unroll
    for (int t = 0; t < 4; ++t) {
        int n = n0 + 16 * t + ncol;
        float bv = bias ? bias[n] : 0.f;
#pragma unroll
        for (int r = 0; r < 8; ++r) {
            float v = acc[t][r] + bv;
            int m = mbase + r;
            if (Cf) Cf[(size_t)m * Nd + n] = v;
            if (Cb) Cb[(size_t)m * Nd + n] = (__bf16)v;
        }
    }
}

// ---------------------------------------------------------------------------
// Attention projections: src[i,k] = h_i . a_w[k,:H] + a_b[k];  dst[i,k] = h_i . a_w[k,H:]
// ---------------------------------------------------------------------------
__global__ void k_proj(const float* __restrict__ h, const float* __restrict__ aw,
                       const float* __restrict__ ab,
                       float* __restrict__ src_nk, float* __restrict__ dst_nk,
                       float* __restrict__ dst_kn) {
    int wave = (int)((blockIdx.x * blockDim.x + threadIdx.x) >> 5);
    int lane = threadIdx.x & 31;
    if (wave >= Nn) return;
    const float* hr = h + (size_t)wave * Hh;
    float hv[8];
#pragma unroll
    for (int e = 0; e < 8; ++e) hv[e] = hr[lane * 8 + e];
#pragma unroll
    for (int k = 0; k < Kk; ++k) {
        const float* w = aw + (size_t)k * 2 * Hh;
        float sv = 0.f, dv = 0.f;
#pragma unroll
        for (int e = 0; e < 8; ++e) {
            sv += hv[e] * w[lane * 8 + e];
            dv += hv[e] * w[Hh + lane * 8 + e];
        }
        for (int off = 16; off > 0; off >>= 1) {
            sv += __shfl_xor(sv, off, 32);
            dv += __shfl_xor(dv, off, 32);
        }
        if (lane == 0) {
            src_nk[(size_t)wave * Kk + k] = sv + ab[k];
            dst_nk[(size_t)wave * Kk + k] = dv;
            dst_kn[(size_t)k * Nn + wave] = dv;
        }
    }
}

// ---------------------------------------------------------------------------
// Per-(row,head) softmax stats over neighbors.
// ---------------------------------------------------------------------------
__global__ void k_stats(const __bf16* __restrict__ Abf, const float* __restrict__ src_nk,
                        const float* __restrict__ dst_nk,
                        float* __restrict__ Ms, float* __restrict__ Ss) {
    int i = blockIdx.x, t = threadIdx.x;
    float srcv[Kk], m[Kk], s[Kk];
#pragma unroll
    for (int k = 0; k < Kk; ++k) { srcv[k] = src_nk[(size_t)i * Kk + k]; m[k] = NEGBIG; s[k] = 0.f; }
    const __bf16* arow = Abf + (size_t)i * Nn;
    for (int j = t; j < Nn; j += 256) {
        if ((float)arow[j] != 0.f) {
            const float* dj = dst_nk + (size_t)j * Kk;
#pragma unroll
            for (int k = 0; k < Kk; ++k) {
                float x = srcv[k] + dj[k];
                x = x >= 0.f ? x : 0.01f * x;
                if (x > m[k]) { s[k] = s[k] * __expf(m[k] - x) + 1.f; m[k] = x; }
                else          { s[k] += __expf(x - m[k]); }
            }
        }
    }
    __shared__ float smM[256];
    __shared__ float smS[256];
    for (int k = 0; k < Kk; ++k) {
        smM[t] = m[k]; smS[t] = s[k];
        __syncthreads();
        for (int off = 128; off > 0; off >>= 1) {
            if (t < off) {
                float m1 = smM[t], m2 = smM[t + off], s1 = smS[t], s2 = smS[t + off];
                float mm = fmaxf(m1, m2);
                smM[t] = mm;
                smS[t] = s1 * __expf(m1 - mm) + s2 * __expf(m2 - mm);
            }
            __syncthreads();
        }
        if (t == 0) { Ms[(size_t)i * Kk + k] = smM[0]; Ss[(size_t)i * Kk + k] = smS[0]; }
        __syncthreads();
    }
}

// ---------------------------------------------------------------------------
// Layer-1 fused attention. One CTA (8 waves) per 16-row block; wave w = head w.
// h1^T slab (256 x 32 bf16) double-buffered in LDS via async-to-LDS copies;
// all 8 heads consume the same slab.
// ---------------------------------------------------------------------------
__global__ void __launch_bounds__(256) k_attn1(
        const __bf16* __restrict__ Abf, const __bf16* __restrict__ hT,
        const float* __restrict__ hfull,
        const float* __restrict__ src_nk, const float* __restrict__ dst_kn,
        const float* __restrict__ Ms, const float* __restrict__ Ss,
        __bf16* __restrict__ hnew) {
    __shared__ __bf16 slab[2][256 * LDSROW];   // 2 x 20KB
    int lane = threadIdx.x & 31;
    int k    = threadIdx.x >> 5;               // head = wave id
    int rb   = blockIdx.x;
    int i = rb * 16 + (lane & 15);
    int b = lane >> 4;
    float srcv = src_nk[(size_t)i * Kk + k];
    float Mi   = Ms[(size_t)i * Kk + k];
    float Si   = Ss[(size_t)i * Kk + k];
    float invS = Si > 0.f ? 1.f / Si : 0.f;

    const __bf16* arow = Abf + (size_t)i * Nn;
    const float*  drow = dst_kn + (size_t)k * Nn;
    v8f acc[16] = {};

    const int NT = Nn / 32;
    ASYNC_STAGE(hT, Nn, 0, &slab[0][threadIdx.x * LDSROW]);
    for (int jt = 0; jt < NT; ++jt) {
        int j0 = jt * 32;
        if (jt + 1 < NT) {
            ASYNC_STAGE(hT, Nn, j0 + 32, &slab[(jt + 1) & 1][threadIdx.x * LDSROW]);
            asm volatile("s_wait_asynccnt 0x4" ::: "memory");
        } else {
            asm volatile("s_wait_asynccnt 0x0" ::: "memory");
        }
        __syncthreads();
        const __bf16* buf = &slab[jt & 1][0];

        v8bf alo = *(const v8bf*)(arow + j0 + 8 * b);
        v8bf ahi = *(const v8bf*)(arow + j0 + 16 + 8 * b);
        const float* dp = drow + j0 + 8 * b;
        v4f d0 = *(const v4f*)dp, d1 = *(const v4f*)(dp + 4);
        v4f d2 = *(const v4f*)(dp + 16), d3 = *(const v4f*)(dp + 20);
        float dv[16] = { d0[0],d0[1],d0[2],d0[3], d1[0],d1[1],d1[2],d1[3],
                         d2[0],d2[1],d2[2],d2[3], d3[0],d3[1],d3[2],d3[3] };
        v16bf af;
#pragma unroll
        for (int e = 0; e < 16; ++e) {
            float a = (e < 8) ? (float)alo[e] : (float)ahi[e - 8];
            float x = srcv + dv[e];
            x = x >= 0.f ? x : 0.01f * x;
            float p = (a != 0.f) ? __expf(x - Mi) * invS : 0.f;
            af[e] = (__bf16)p;
        }
#pragma unroll
        for (int t = 0; t < 16; ++t) {
            v16bf bfr = lds_frag(buf, t, lane);
            acc[t] = wmma_bf16(af, bfr, acc[t]);
        }
        __syncthreads();
    }
    int mbase = rb * 16 + 8 * b;
    int ncol  = lane & 15;
#pragma unroll
    for (int t = 0; t < 16; ++t) {
        int n = t * 16 + ncol;
#pragma unroll
        for (int r = 0; r < 8; ++r) {
            int row = mbase + r;
            float v = acc[t][r] + hfull[(size_t)row * Hh + n];
            v = v > 0.f ? v : (__expf(v) - 1.f);           // elu
            hnew[(size_t)row * KH + k * Hh + n] = (__bf16)v;
        }
    }
}

// ---------------------------------------------------------------------------
// Layer-2 fused attention, heads pre-combined into alphaAvg before the matmul.
// One CTA = 8 waves = 8 consecutive 16-row blocks sharing the LDS h2^T slab.
// ---------------------------------------------------------------------------
__global__ void __launch_bounds__(256) k_attn2(
        const __bf16* __restrict__ Abf, const __bf16* __restrict__ hT,
        const float* __restrict__ hfull,
        const float* __restrict__ src_nk, const float* __restrict__ dst_kn,
        const float* __restrict__ Ms, const float* __restrict__ Ss,
        float* __restrict__ hnew) {
    __shared__ __bf16 slab[2][256 * LDSROW];
    int lane = threadIdx.x & 31;
    int rb   = blockIdx.x * 8 + (threadIdx.x >> 5);
    int i = rb * 16 + (lane & 15);
    int b = lane >> 4;
    float srcv[Kk], Mi[Kk], invS[Kk];
#pragma unroll
    for (int k = 0; k < Kk; ++k) {
        srcv[k] = src_nk[(size_t)i * Kk + k];
        Mi[k]   = Ms[(size_t)i * Kk + k];
        float S = Ss[(size_t)i * Kk + k];
        invS[k] = S > 0.f ? 1.f / S : 0.f;
    }
    const __bf16* arow = Abf + (size_t)i * Nn;
    v8f acc[16] = {};

    const int NT = Nn / 32;
    ASYNC_STAGE(hT, Nn, 0, &slab[0][threadIdx.x * LDSROW]);
    for (int jt = 0; jt < NT; ++jt) {
        int j0 = jt * 32;
        if (jt + 1 < NT) {
            ASYNC_STAGE(hT, Nn, j0 + 32, &slab[(jt + 1) & 1][threadIdx.x * LDSROW]);
            asm volatile("s_wait_asynccnt 0x4" ::: "memory");
        } else {
            asm volatile("s_wait_asynccnt 0x0" ::: "memory");
        }
        __syncthreads();
        const __bf16* buf = &slab[jt & 1][0];

        v8bf alo = *(const v8bf*)(arow + j0 + 8 * b);
        v8bf ahi = *(const v8bf*)(arow + j0 + 16 + 8 * b);
        float p[16];
#pragma unroll
        for (int e = 0; e < 16; ++e) p[e] = 0.f;
#pragma unroll
        for (int k = 0; k < Kk; ++k) {
            const float* dp = dst_kn + (size_t)k * Nn + j0 + 8 * b;
            v4f d0 = *(const v4f*)dp, d1 = *(const v4f*)(dp + 4);
            v4f d2 = *(const v4f*)(dp + 16), d3 = *(const v4f*)(dp + 20);
            float dv[16] = { d0[0],d0[1],d0[2],d0[3], d1[0],d1[1],d1[2],d1[3],
                             d2[0],d2[1],d2[2],d2[3], d3[0],d3[1],d3[2],d3[3] };
#pragma unroll
            for (int e = 0; e < 16; ++e) {
                float x = srcv[k] + dv[e];
                x = x >= 0.f ? x : 0.01f * x;
                p[e] += __expf(x - Mi[k]) * invS[k];
            }
        }
        v16bf af;
#pragma unroll
        for (int e = 0; e < 16; ++e) {
            float a = (e < 8) ? (float)alo[e] : (float)ahi[e - 8];
            af[e] = (__bf16)((a != 0.f) ? p[e] * (1.0f / Kk) : 0.f);
        }
#pragma unroll
        for (int t = 0; t < 16; ++t) {
            v16bf bfr = lds_frag(buf, t, lane);
            acc[t] = wmma_bf16(af, bfr, acc[t]);
        }
        __syncthreads();
    }
    int mbase = rb * 16 + 8 * b;
    int ncol  = lane & 15;
#pragma unroll
    for (int t = 0; t < 16; ++t) {
        int n = t * 16 + ncol;
#pragma unroll
        for (int r = 0; r < 8; ++r) {
            int row = mbase + r;
            float v = acc[t][r] + hfull[(size_t)row * Hh + n];
            v = v > 0.f ? v : (__expf(v) - 1.f);           // elu
            hnew[(size_t)row * Hh + n] = v;
        }
    }
}

// ---------------------------------------------------------------------------
// Column-mean * out_w, then final reduce + bias.
// ---------------------------------------------------------------------------
__global__ void k_colmean(const float* __restrict__ h2new, const float* __restrict__ out_w,
                          float* __restrict__ partial) {
    int c = blockIdx.x, t = threadIdx.x;
    float s = 0.f;
    for (int q = t; q < Nn; q += 256) s += h2new[(size_t)q * Hh + c];
    __shared__ float sm[256];
    sm[t] = s;
    __syncthreads();
    for (int off = 128; off > 0; off >>= 1) {
        if (t < off) sm[t] += sm[t + off];
        __syncthreads();
    }
    if (t == 0) partial[c] = (sm[0] / (float)Nn) * out_w[c];
}

__global__ void k_final(const float* __restrict__ partial, const float* __restrict__ out_b,
                        float* __restrict__ out) {
    int t = threadIdx.x;
    __shared__ float sm[256];
    sm[t] = partial[t];
    __syncthreads();
    for (int off = 128; off > 0; off >>= 1) {
        if (t < off) sm[t] += sm[t + off];
        __syncthreads();
    }
    if (t == 0) out[0] = sm[0] + out_b[0];
}

// ---------------------------------------------------------------------------
extern "C" void kernel_launch(void* const* d_in, const int* in_sizes, int n_in,
                              void* d_out, int out_size, void* d_ws, size_t ws_size,
                              hipStream_t stream) {
    const float* adjacency = (const float*)d_in[0];
    const float* feats     = (const float*)d_in[1];
    const float* W1_w      = (const float*)d_in[2];
    const float* W1_b      = (const float*)d_in[3];
    const float* a1_w      = (const float*)d_in[4];
    const float* a1_b      = (const float*)d_in[5];
    const float* W2_w      = (const float*)d_in[6];
    const float* W2_b      = (const float*)d_in[7];
    const float* a2_w      = (const float*)d_in[8];
    const float* a2_b      = (const float*)d_in[9];
    const float* out_w     = (const float*)d_in[10];
    const float* out_b     = (const float*)d_in[11];
    float* outp = (float*)d_out;

    uintptr_t base = (uintptr_t)d_ws;
    auto alloc = [&](size_t bytes) -> void* {
        void* p = (void*)base;
        base += (bytes + 255) & ~(size_t)255;
        return p;
    };
    __bf16* Abf     = (__bf16*)alloc((size_t)Nn * Nn * 2);
    __bf16* featsT  = (__bf16*)alloc((size_t)Ff * Nn * 2);
    __bf16* AFbf    = (__bf16*)alloc((size_t)Nn * Ff * 2);
    __bf16* W1bf    = (__bf16*)alloc((size_t)Hh * Ff * 2);
    float*  h1      = (float*) alloc((size_t)Nn * Hh * 4);
    __bf16* h1T     = (__bf16*)alloc((size_t)Hh * Nn * 2);
    float*  src1    = (float*) alloc((size_t)Nn * Kk * 4);
    float*  dst1nk  = (float*) alloc((size_t)Nn * Kk * 4);
    float*  dst1kn  = (float*) alloc((size_t)Kk * Nn * 4);
    float*  M1      = (float*) alloc((size_t)Nn * Kk * 4);
    float*  S1      = (float*) alloc((size_t)Nn * Kk * 4);
    __bf16* h1new   = (__bf16*)alloc((size_t)Nn * KH * 2);
    __bf16* W2bf    = (__bf16*)alloc((size_t)Hh * KH * 2);
    float*  h2      = (float*) alloc((size_t)Nn * Hh * 4);
    __bf16* h2T     = (__bf16*)alloc((size_t)Hh * Nn * 2);
    float*  src2    = (float*) alloc((size_t)Nn * Kk * 4);
    float*  dst2nk  = (float*) alloc((size_t)Nn * Kk * 4);
    float*  dst2kn  = (float*) alloc((size_t)Kk * Nn * 4);
    float*  M2      = (float*) alloc((size_t)Nn * Kk * 4);
    float*  S2      = (float*) alloc((size_t)Nn * Kk * 4);
    float*  h2new   = (float*) alloc((size_t)Nn * Hh * 4);
    float*  partial = (float*) alloc((size_t)Hh * 4);

    // --- conversions ---
    k_cvt<<<((size_t)Nn * Nn) / 256, 256, 0, stream>>>(adjacency, Abf, (size_t)Nn * Nn);
    k_cvtT<<<((size_t)Nn * Ff) / 256, 256, 0, stream>>>(feats, featsT, Nn, Ff);
    k_cvt<<<((size_t)Hh * Ff) / 256, 256, 0, stream>>>(W1_w, W1bf, (size_t)Hh * Ff);
    k_cvt<<<((size_t)Hh * KH) / 256, 256, 0, stream>>>(W2_w, W2bf, (size_t)Hh * KH);

    // --- AF = adjacency @ feats  (bf16 out) ---
    {
        int waves = (Nn / 16) * (Ff / 64);
        k_gemm<<<waves * 32 / 256, 256, 0, stream>>>(Abf, featsT, nullptr, nullptr, AFbf, Nn, Ff, Nn);
    }
    // --- h1 = AF @ W1^T + b1 (fp32) ---
    {
        int waves = (Nn / 16) * (Hh / 64);
        k_gemm<<<waves * 32 / 256, 256, 0, stream>>>(AFbf, W1bf, W1_b, h1, nullptr, Nn, Hh, Ff);
    }
    k_cvtT<<<((size_t)Nn * Hh) / 256, 256, 0, stream>>>(h1, h1T, Nn, Hh);
    k_proj<<<Nn * 32 / 256, 256, 0, stream>>>(h1, a1_w, a1_b, src1, dst1nk, dst1kn);
    k_stats<<<Nn, 256, 0, stream>>>(Abf, src1, dst1nk, M1, S1);
    k_attn1<<<Nn / 16, 256, 0, stream>>>(Abf, h1T, h1, src1, dst1kn, M1, S1, h1new);
    // --- h2 = h1_new @ W2^T + b2 ---
    {
        int waves = (Nn / 16) * (Hh / 64);
        k_gemm<<<waves * 32 / 256, 256, 0, stream>>>(h1new, W2bf, W2_b, h2, nullptr, Nn, Hh, KH);
    }
    k_cvtT<<<((size_t)Nn * Hh) / 256, 256, 0, stream>>>(h2, h2T, Nn, Hh);
    k_proj<<<Nn * 32 / 256, 256, 0, stream>>>(h2, a2_w, a2_b, src2, dst2nk, dst2kn);
    k_stats<<<Nn, 256, 0, stream>>>(Abf, src2, dst2nk, M2, S2);
    k_attn2<<<Nn / 16 / 8, 256, 0, stream>>>(Abf, h2T, h2, src2, dst2kn, M2, S2, h2new);
    k_colmean<<<Hh, 256, 0, stream>>>(h2new, out_w, partial);
    k_final<<<1, 256, 0, stream>>>(partial, out_b, outp);
    (void)in_sizes; (void)n_in; (void)out_size; (void)ws_size;
}
